// SNN_Georg_et_al_Equalizer_69526930587732
// MI455X (gfx1250) — matrix operand order repaired and checked by reading, not compiled, not run on hardware
//
#include <hip/hip_runtime.h>

// ---------------------------------------------------------------------------
// SNN LIF scan, fused persistent kernel for MI455X (gfx1250, wave32, WMMA).
// T=512, B=256, I=64, H=256, O=4.
// Grid: 16 workgroups (batch tiles of 16), 512 threads = 16 waves each.
// Wave w handles hidden columns [16w,16w+16); state (v1,i1) lives in the
// WMMA C/D accumulator layout (8 VGPRs per lane = 8 batch rows).
// Per step per wave: 2x v_wmma_f32_16x16x32_f16 (input GEMM, K=64) +
// 1x v_wmma_f32_16x16x32_f16 (spike GEMM, K=16 valid, rest zero).
// x_t tiles are prefetched DEPTH-1 steps ahead with GLOBAL_LOAD_ASYNC_TO_LDS
// (ASYNCcnt) so the sequential scan never waits on HBM latency.
// ---------------------------------------------------------------------------

typedef __attribute__((ext_vector_type(16))) _Float16 v16h;
typedef __attribute__((ext_vector_type(8)))  float    v8f;
typedef __attribute__((ext_vector_type(2)))  int      v2i_t;

#define T_STEPS 512
#define BATCH   256
#define IN_DIM  64
#define HID     256
#define OUT_DIM 4
#define DT      0.001f
#define XSTRF   66   // padded f32 row stride: banks (2m+k)%64, conflict-free
#define DEPTH   4    // async prefetch ring depth

#if defined(__has_builtin)
#  if __has_builtin(__builtin_amdgcn_global_load_async_to_lds_b64) && \
      __has_builtin(__builtin_amdgcn_s_wait_asynccnt)
#    define HAVE_ASYNC 1
#  endif
#endif
#ifndef HAVE_ASYNC
#  define HAVE_ASYNC 0
#endif

#if HAVE_ASYNC
typedef __attribute__((address_space(1))) v2i_t* gptr_b64;
typedef __attribute__((address_space(3))) v2i_t* lptr_b64;
#endif

__global__ void init_acc_kernel(unsigned* acc) {
  if (threadIdx.x == 0 && blockIdx.x == 0) *acc = 0u;
}

__global__ void finalize_kernel(const unsigned* acc, float* out) {
  if (threadIdx.x == 0 && blockIdx.x == 0) {
    out[BATCH * OUT_DIM] =
        (float)(*acc) / ((float)T_STEPS * (float)BATCH * (float)HID);
  }
}

// Stage one x_t tile (16 batch x 64 in, fp32) into an LDS ring slot.
// 512 threads x 2 floats (8 bytes) = full 4KB tile.
static __device__ __forceinline__ void stage_issue(const float* __restrict__ x,
                                                   int b0, int t,
                                                   float* dst /*LDS*/) {
  const int tid = threadIdx.x;
  const int e  = tid * 2;
  const int sm = e >> 6, sk = e & 63;
  const float* g = x + ((size_t)t * BATCH + b0 + sm) * IN_DIM + sk;
#if HAVE_ASYNC
  __builtin_amdgcn_global_load_async_to_lds_b64(
      (gptr_b64)g, (lptr_b64)(dst + sm * XSTRF + sk), 0, 0);
#else
  const float2 v = *(const float2*)g;
  dst[sm * XSTRF + sk]     = v.x;
  dst[sm * XSTRF + sk + 1] = v.y;
#endif
}

__global__ void __launch_bounds__(512)
snn_scan_kernel(const float* __restrict__ x,            // [T,B,I]
                const float* __restrict__ W_in,         // [H,I]
                const float* __restrict__ b_in,         // [H]
                const float* __restrict__ W_out,        // [O,H]
                const float* __restrict__ b_out,        // [O]
                const float* __restrict__ v_th,         // [H]
                const float* __restrict__ v_reset,      // [H]
                const float* __restrict__ v_leak1p,     // scalar
                const float* __restrict__ tau_mem_inv1, // [H]
                const float* __restrict__ tau_syn_inv1, // [H]
                const float* __restrict__ v_leak2p,     // scalar
                const float* __restrict__ tau_mem_inv2, // [O]
                const float* __restrict__ tau_syn_inv2, // [O]
                float* __restrict__ out,                // [B*O] (+1 elsewhere)
                unsigned* __restrict__ spike_acc)
{
  __shared__ float    xbuf[DEPTH][16 * XSTRF];  // async x_t ring, f32
  __shared__ _Float16 zbuf[16][16][16];         // [wave][batch_m][hid_col]
  __shared__ float    part[16][16][OUT_DIM];    // [wave][batch_m][out_j]
  __shared__ unsigned wg_spk;

  const int tid = threadIdx.x;
  const int w   = tid >> 5;        // wave id (16 waves)
  const int l   = tid & 31;        // lane (wave32)
  const int g   = blockIdx.x;      // batch tile
  const int b0  = g * 16;
  const int h0  = w * 16;
  const int n   = l & 15;          // B/C/D column within tile
  const int hi  = (l >= 16) ? 1 : 0;

  if (tid == 0) wg_spk = 0u;

  // ---- per-lane hidden-column parameters (column h = h0 + n) ----
  const int   h      = h0 + n;
  const float p_vth  = v_th[h];
  const float p_vrst = v_reset[h];
  const float p_bin  = b_in[h];
  const float dt_tmi = DT * tau_mem_inv1[h];
  const float dt_tsi = DT * tau_syn_inv1[h];
  const float vleak1 = v_leak1p[0];

  // ---- W_in B-fragments (B is 32x16 K-major: lanes 0-15 hold K=0..7,16..23,
  //      lanes 16-31 hold K=8..15,24..31; matches A table order) ----
  v16h bw0, bw1;
#pragma unroll
  for (int hh = 0; hh < 16; ++hh) {
    const int k = (hh & 7) + ((hh >> 3) << 4) + (hi ? 8 : 0);
    bw0[hh] = (_Float16)W_in[h * IN_DIM + k];
    bw1[hh] = (_Float16)W_in[h * IN_DIM + k + 32];
  }

  // ---- W_out B-fragment: K = local hidden 0..15 valid (rest zero),
  //      column n<4 valid. B[k][n] = W_out[n*H + h0 + k] ----
  v16h bo;
#pragma unroll
  for (int hh = 0; hh < 16; ++hh) {
    float val = 0.0f;
    if (hh < 8 && n < OUT_DIM) val = W_out[n * HID + h0 + hh + (hi ? 8 : 0)];
    bo[hh] = (_Float16)val;
  }

  // ---- persistent per-lane state: 8 batch rows x 1 hidden column ----
  float v1[8], i1[8];
#pragma unroll
  for (int r = 0; r < 8; ++r) { v1[r] = 0.0f; i1[r] = 0.0f; }
  int scnt = 0;

  // ---- output-layer threads (tid<64): (batch om, out oj) ----
  const int om = tid >> 2;
  const int oj = tid & 3;
  float v2 = 0.0f, i2 = 0.0f, vmax = -3.402823466e38f;
  float o_dt_tmi = 0.0f, o_dt_tsi = 0.0f, o_bout = 0.0f, vleak2 = 0.0f;
  if (tid < 64) {
    o_dt_tmi = DT * tau_mem_inv2[oj];
    o_dt_tsi = DT * tau_syn_inv2[oj];
    o_bout   = b_out[oj];
    vleak2   = v_leak2p[0];
  }

#if HAVE_ASYNC
  // prologue: fill the prefetch pipeline (tiles 0 .. DEPTH-2)
#pragma unroll
  for (int s = 0; s < DEPTH - 1; ++s) stage_issue(x, b0, s, xbuf[s]);
#endif

  for (int t = 0; t < T_STEPS; ++t) {
#if HAVE_ASYNC
    {
      // issue prefetch for tile t+DEPTH-1 into ring slot (t-1)%DEPTH.
      // Clamp the tile index near the end (uniform issue keeps ASYNCcnt
      // accounting exact and avoids EXEC divergence around the async op).
      int tp = t + DEPTH - 1;
      if (tp >= T_STEPS) tp = T_STEPS - 1;
      stage_issue(x, b0, tp, xbuf[(t + DEPTH - 1) % DEPTH]);
      // loads complete in order: <= DEPTH-1 outstanding => tile t resident
      __builtin_amdgcn_s_wait_asynccnt(DEPTH - 1);
    }
    __syncthreads();                    // tile t visible to all waves;
                                        // also fences part[] reuse
    const float* xb = xbuf[t % DEPTH];
#else
    __syncthreads();                    // fence xbuf/part reuse
    stage_issue(x, b0, t, xbuf[0]);
    __syncthreads();
    const float* xb = xbuf[0];
#endif

    // ---- A-fragments from LDS (A 16x32 f16 layout):
    //      VGPR v: K = 2*(v&3) + 16*(v>>2) + (lane>=16 ? 8 : 0) ----
    const int am = n;   // A row = batch row = lane%16
    v16h a0, a1;
#pragma unroll
    for (int v = 0; v < 8; ++v) {
      const int k = ((v & 3) << 1) + ((v >> 2) << 4) + (hi ? 8 : 0);
      a0[2 * v]     = (_Float16)xb[am * XSTRF + k];
      a0[2 * v + 1] = (_Float16)xb[am * XSTRF + k + 1];
      a1[2 * v]     = (_Float16)xb[am * XSTRF + k + 32];
      a1[2 * v + 1] = (_Float16)xb[am * XSTRF + k + 33];
    }

    v8f cur = {};
    cur = __builtin_amdgcn_wmma_f32_16x16x32_f16(false, a0, false, bw0,
                                                 (short)0, cur, false, false);
    cur = __builtin_amdgcn_wmma_f32_16x16x32_f16(false, a1, false, bw1,
                                                 (short)0, cur, false, false);

    // ---- LIF dynamics on accumulator layout; spikes -> zbuf ----
#pragma unroll
    for (int r = 0; r < 8; ++r) {
      const float vv    = v1[r];
      const float ii    = i1[r];
      const float v_dec = vv + dt_tmi * ((vleak1 - vv) + ii);
      const float i_dec = ii - dt_tsi * ii;
      const bool  zb    = (v_dec - p_vth) > 0.0f;
      v1[r] = zb ? p_vrst : v_dec;
      i1[r] = i_dec + cur[r] + p_bin;
      scnt += zb ? 1 : 0;
      zbuf[w][r + (hi ? 8 : 0)][n] = (_Float16)(zb ? 1.0f : 0.0f);
    }

    // ---- spike GEMM: A = z (16 batch x 16 hid, K padded to 32) ----
    v16h az = {};   // halves 8..15 stay zero (K=16..31)
#pragma unroll
    for (int v = 0; v < 4; ++v) {
      const int k = (v << 1) + (hi ? 8 : 0);
      az[2 * v]     = zbuf[w][am][k];
      az[2 * v + 1] = zbuf[w][am][k + 1];
    }
    v8f po = {};
    po = __builtin_amdgcn_wmma_f32_16x16x32_f16(false, az, false, bo,
                                                (short)0, po, false, false);
    if (n < OUT_DIM) {
#pragma unroll
      for (int r = 0; r < 8; ++r)
        part[w][r + (hi ? 8 : 0)][n] = po[r];
    }
    __syncthreads();

    // ---- output LI update (64 threads: 16 batch x 4 out) ----
    if (tid < 64) {
      float o = o_bout;
#pragma unroll
      for (int ww = 0; ww < 16; ++ww) o += part[ww][om][oj];
      const float v2n = v2 + o_dt_tmi * ((vleak2 - v2) + i2);
      const float i2n = i2 - o_dt_tsi * i2 + o;
      v2 = v2n;
      i2 = i2n;
      vmax = fmaxf(vmax, v2n);
    }
  }

  if (tid < 64) out[(b0 + om) * OUT_DIM + oj] = vmax;

  // deterministic spike-count reduction (integers, exact)
  atomicAdd(&wg_spk, (unsigned)scnt);
  __syncthreads();
  if (tid == 0) atomicAdd(spike_acc, wg_spk);
}

extern "C" void kernel_launch(void* const* d_in, const int* in_sizes, int n_in,
                              void* d_out, int out_size, void* d_ws,
                              size_t ws_size, hipStream_t stream) {
  (void)in_sizes; (void)n_in; (void)out_size; (void)ws_size;

  const float* x    = (const float*)d_in[0];
  const float* W_in = (const float*)d_in[1];
  const float* b_in = (const float*)d_in[2];
  const float* W_out= (const float*)d_in[3];
  const float* b_out= (const float*)d_in[4];
  // d_in[5] = alpha (surrogate-gradient only; unused in forward)
  const float* v_th        = (const float*)d_in[6];
  const float* v_reset     = (const float*)d_in[7];
  const float* v_leak1     = (const float*)d_in[8];
  const float* tau_mem_inv1= (const float*)d_in[9];
  const float* tau_syn_inv1= (const float*)d_in[10];
  const float* v_leak2     = (const float*)d_in[11];
  const float* tau_mem_inv2= (const float*)d_in[12];
  const float* tau_syn_inv2= (const float*)d_in[13];

  float*    out = (float*)d_out;
  unsigned* acc = (unsigned*)d_ws;

  init_acc_kernel<<<1, 1, 0, stream>>>(acc);
  snn_scan_kernel<<<BATCH / 16, 512, 0, stream>>>(
      x, W_in, b_in, W_out, b_out, v_th, v_reset, v_leak1,
      tau_mem_inv1, tau_syn_inv1, v_leak2, tau_mem_inv2, tau_syn_inv2,
      out, acc);
  finalize_kernel<<<1, 1, 0, stream>>>(acc, out);
}
